// Seq2SeqModel_69020124447191
// MI455X (gfx1250) — matrix operand order repaired and checked by reading, not compiled, not run on hardware
//
#include <hip/hip_runtime.h>
#include <hip/hip_bf16.h>
#include <math.h>

// ---------------- problem constants (match reference) ----------------
#define BB    64
#define SIN   128
#define SOUT  32
#define EE    256
#define HH    1024
#define FEAT  256
#define YD    256
#define DIN   (FEAT + YD)   // 512
#define G3H   (3 * HH)      // 3072

typedef __bf16 bf16;
typedef __attribute__((ext_vector_type(8)))  __bf16 v8bf;
typedef __attribute__((ext_vector_type(16))) __bf16 v16bf;
typedef __attribute__((ext_vector_type(8)))  float  v8f;

__device__ __forceinline__ float sigmoidf_(float x) { return 1.0f / (1.0f + __expf(-x)); }

// =====================================================================
// WMMA GEMM (bf16 in, f32 out):  C[M,N] = A[M,K] @ W[N,K]^T + bias[N]
// One wave32 computes a 16(M) x 64(N) block: one A fragment reused by
// four B fragments -> 4 independent v_wmma_f32_16x16x32_bf16 per K-step.
// Fragment layouts used here (same contiguous K order for A and B):
//   A (16x32 bf16): lane<16 -> M=lane,    K = k+0..k+15 (contiguous)
//                   lane>=16-> M=lane-16, K = k+16..k+31
//   B (32x16 bf16): lane<16 -> N=lane,    K = k+0..k+15 (contiguous)
//                   lane>=16-> N=lane-16, K = k+16..k+31
//   C/D (16x16 f32): vgpr r, lane L -> row = r + (L<16?0:8), col = L&15
// =====================================================================
__global__ void gemm_bf16_wmma4(const bf16* __restrict__ A,
                                const bf16* __restrict__ W,
                                const float* __restrict__ bias,
                                float* __restrict__ C,
                                int M, int N, int K)
{
    const int wavesPerBlock = blockDim.x >> 5;
    const int waveId = blockIdx.x * wavesPerBlock + (threadIdx.x >> 5);
    const int lane   = threadIdx.x & 31;
    const int nsup   = N >> 6;                 // 64-wide N supertiles
    const int mtiles = M >> 4;
    if (waveId >= mtiles * nsup) return;       // wave-uniform exit
    const int tileM = (waveId / nsup) << 4;
    const int tileN = (waveId % nsup) << 6;

    const int mA  = tileM + (lane & 15);
    const int kA0 = (lane < 16) ? 0 : 16;
    const int kB0 = (lane < 16) ? 0 : 16;

    const bf16* __restrict__ arow = A + (size_t)mA * K + kA0;
    const bf16* __restrict__ b0   = W + (size_t)(tileN + (lane & 15)) * K + kB0;
    const bf16* __restrict__ b1   = b0 + (size_t)16 * K;
    const bf16* __restrict__ b2   = b0 + (size_t)32 * K;
    const bf16* __restrict__ b3   = b0 + (size_t)48 * K;

    v8f acc0 = {}, acc1 = {}, acc2 = {}, acc3 = {};
    for (int k = 0; k < K; k += 32) {
        // A fragment: two 16B loads (K..K+7 | K+8..K+15 per lane half)
        v8bf alo = *(const v8bf*)(arow + k);
        v8bf ahi = *(const v8bf*)(arow + k + 8);
        v16bf af = __builtin_shufflevector(alo, ahi,
                      0,1,2,3,4,5,6,7,8,9,10,11,12,13,14,15);
        // B fragments: one 32B load each (16 contiguous bf16 along K)
        v16bf bf0 = *(const v16bf*)(b0 + k);
        v16bf bf1 = *(const v16bf*)(b1 + k);
        v16bf bf2 = *(const v16bf*)(b2 + k);
        v16bf bf3 = *(const v16bf*)(b3 + k);
        // prefetch next K-block of the weight stream (L2 -> L0)
        __builtin_prefetch((const void*)(b0 + k + 32), 0, 1);

        acc0 = __builtin_amdgcn_wmma_f32_16x16x32_bf16(
                   false, af, false, bf0, (short)0, acc0, false, false);
        acc1 = __builtin_amdgcn_wmma_f32_16x16x32_bf16(
                   false, af, false, bf1, (short)0, acc1, false, false);
        acc2 = __builtin_amdgcn_wmma_f32_16x16x32_bf16(
                   false, af, false, bf2, (short)0, acc2, false, false);
        acc3 = __builtin_amdgcn_wmma_f32_16x16x32_bf16(
                   false, af, false, bf3, (short)0, acc3, false, false);
    }

    const int rowOff   = (lane < 16) ? 0 : 8;
    const int colBase  = tileN + (lane & 15);
#define STORE_TILE(ACC, TT)                                                  \
    {                                                                        \
        const int col = colBase + 16 * (TT);                                 \
        const float bv = bias ? bias[col] : 0.0f;                            \
        _Pragma("unroll")                                                    \
        for (int r = 0; r < 8; ++r)                                          \
            C[(size_t)(tileM + rowOff + r) * N + col] = (ACC)[r] + bv;       \
    }
    STORE_TILE(acc0, 0)
    STORE_TILE(acc1, 1)
    STORE_TILE(acc2, 2)
    STORE_TILE(acc3, 3)
#undef STORE_TILE
}

// ---------------- f32 -> bf16 conversion (weights / one-shot) ---------
__global__ void cvt_bf16(const float* __restrict__ src, bf16* __restrict__ dst, int n)
{
    int i = blockIdx.x * blockDim.x + threadIdx.x;
    if (i < n) dst[i] = (bf16)src[i];
}

// ---------------- GRU elementwise update (PyTorch gate order r,z,n) ---
// Writes fp32 master state + bf16 copy (GEMM input for next step),
// and optional fp32 / bf16 sequence outputs.
__global__ void gru_update(const float* __restrict__ gi, int giStrideB,
                           const float* __restrict__ gh,
                           const float* __restrict__ h_in,
                           float* __restrict__ h_out,
                           bf16*  __restrict__ h_out_bf,
                           float* __restrict__ seq_out,
                           bf16*  __restrict__ seq_bf, int seqStrideB)
{
    int i = blockIdx.x * blockDim.x + threadIdx.x;
    if (i >= BB * HH) return;
    int b = i >> 10;            // H = 1024
    int j = i & (HH - 1);
    const float* gib = gi + (size_t)b * giStrideB;
    const float* ghb = gh + (size_t)b * G3H;
    float r = sigmoidf_(gib[j]          + ghb[j]);
    float z = sigmoidf_(gib[HH + j]     + ghb[HH + j]);
    float n = tanhf    (gib[2 * HH + j] + r * ghb[2 * HH + j]);
    float h = (1.0f - z) * n + z * h_in[i];
    h_out[i]    = h;
    h_out_bf[i] = (bf16)h;
    if (seq_out) seq_out[(size_t)b * seqStrideB + j] = h;
    if (seq_bf)  seq_bf [(size_t)b * seqStrideB + j] = (bf16)h;
}

// ---------------- embedding gather (bf16 out: feeds GEMM only) --------
__global__ void embed_kernel(const int* __restrict__ x,
                             const float* __restrict__ emb,
                             bf16* __restrict__ xe)
{
    int i = blockIdx.x * blockDim.x + threadIdx.x;
    if (i >= BB * SIN * EE) return;
    int e  = i & (EE - 1);
    int bt = i / EE;
    xe[i] = (bf16)emb[(size_t)x[bt] * EE + e];
}

// ---------------- decoder input build (teacher forcing shift) ---------
__global__ void build_dec_in(const float* __restrict__ mp,
                             const float* __restrict__ y,
                             bf16* __restrict__ dec_in)
{
    int i = blockIdx.x * blockDim.x + threadIdx.x;
    if (i >= BB * SOUT * DIN) return;
    int d  = i % DIN;
    int bt = i / DIN;
    int t  = bt % SOUT;
    int b  = bt / SOUT;
    float v;
    if (d < FEAT) v = mp[((size_t)b * SOUT + t) * FEAT + d];
    else          v = (t == 0) ? 0.0f : y[((size_t)b * SOUT + (t - 1)) * YD + (d - FEAT)];
    dec_in[i] = (bf16)v;
}

// ---------------- attention scores: one wave per (b, s) ---------------
__global__ void attn_scores(const float* __restrict__ enc_out,
                            const float* __restrict__ h1,
                            float* __restrict__ scores)
{
    int wid  = blockIdx.x * (blockDim.x >> 5) + (threadIdx.x >> 5);
    int lane = threadIdx.x & 31;
    if (wid >= BB * SIN) return;
    int b = wid / SIN, s = wid % SIN;
    const float* e = enc_out + ((size_t)b * SIN + s) * HH;
    const float* h = h1 + (size_t)b * HH;
    float acc = 0.0f;
    for (int j = lane; j < HH; j += 32) acc += e[j] * h[j];
#pragma unroll
    for (int off = 16; off; off >>= 1) acc += __shfl_down(acc, off, 32);
    if (lane == 0) scores[wid] = acc;
}

// ---------------- softmax over SIN=128, one block per batch row -------
__global__ void softmax128(const float* __restrict__ scores,
                           float* __restrict__ alpha)
{
    __shared__ float sm[SIN];
    int b = blockIdx.x, s = threadIdx.x;
    float v = scores[b * SIN + s];
    sm[s] = v; __syncthreads();
    for (int off = 64; off; off >>= 1) {
        if (s < off) sm[s] = fmaxf(sm[s], sm[s + off]);
        __syncthreads();
    }
    float mx = sm[0]; __syncthreads();
    float e = __expf(v - mx);
    sm[s] = e; __syncthreads();
    for (int off = 64; off; off >>= 1) {
        if (s < off) sm[s] += sm[s + off];
        __syncthreads();
    }
    alpha[b * SIN + s] = e / sm[0];
}

// ---------------- attn context + concat [attn, h1] (bf16: GEMM A) -----
__global__ void attn_apply(const float* __restrict__ enc_out,
                           const float* __restrict__ alpha,
                           const float* __restrict__ h1,
                           bf16* __restrict__ ctx)   // [B, 2H] bf16
{
    int i = blockIdx.x * blockDim.x + threadIdx.x;
    if (i >= BB * HH) return;
    int b = i >> 10, j = i & (HH - 1);
    const float* e = enc_out + (size_t)b * SIN * HH + j;
    const float* a = alpha + b * SIN;
    float acc = 0.0f;
    for (int s = 0; s < SIN; ++s) acc += a[s] * e[(size_t)s * HH];
    ctx[(size_t)b * 2 * HH + j]      = (bf16)acc;
    ctx[(size_t)b * 2 * HH + HH + j] = (bf16)h1[i];
}

// ---------------- log-softmax loss: one block(256) per batch row ------
__global__ void loss_kernel(const float* __restrict__ logits,
                            const float* __restrict__ y, int t,
                            float* __restrict__ loss_acc)
{
    __shared__ float sm[YD];
    int b = blockIdx.x, d = threadIdx.x;
    float v = logits[b * YD + d];
    sm[d] = v; __syncthreads();
    for (int off = 128; off; off >>= 1) {
        if (d < off) sm[d] = fmaxf(sm[d], sm[d + off]);
        __syncthreads();
    }
    float mx = sm[0]; __syncthreads();
    float e = __expf(v - mx);
    sm[d] = e; __syncthreads();
    for (int off = 128; off; off >>= 1) {
        if (d < off) sm[d] += sm[d + off];
        __syncthreads();
    }
    float lse = mx + __logf(sm[0]);
    __syncthreads();
    float yt = y[((size_t)b * SOUT + t) * YD + d];
    sm[d] = -(v - lse) * yt; __syncthreads();
    for (int off = 128; off; off >>= 1) {
        if (d < off) sm[d] += sm[d + off];
        __syncthreads();
    }
    if (d == 0) atomicAdd(loss_acc, sm[0]);
}

// ---------------- misc ------------------------------------------------
__global__ void zero_kernel(float* __restrict__ p, int n)
{
    int i = blockIdx.x * blockDim.x + threadIdx.x;
    if (i < n) p[i] = 0.0f;
}

__global__ void zero_bf16(bf16* __restrict__ p, int n)
{
    int i = blockIdx.x * blockDim.x + threadIdx.x;
    if (i < n) p[i] = (bf16)0.0f;
}

__global__ void sum_all(const float* __restrict__ y, int n, float* __restrict__ out)
{
    __shared__ float sm[256];
    float acc = 0.0f;
    for (int i = blockIdx.x * blockDim.x + threadIdx.x; i < n; i += gridDim.x * blockDim.x)
        acc += y[i];
    sm[threadIdx.x] = acc; __syncthreads();
    for (int off = 128; off; off >>= 1) {
        if (threadIdx.x < off) sm[threadIdx.x] += sm[threadIdx.x + off];
        __syncthreads();
    }
    if (threadIdx.x == 0) atomicAdd(out, sm[0]);
}

__global__ void finalize(const float* __restrict__ loss,
                         const float* __restrict__ ysum,
                         float* __restrict__ out)
{
    if (blockIdx.x == 0 && threadIdx.x == 0) out[0] = loss[0] / ysum[0];
}

// =====================================================================
// host-side orchestration
// =====================================================================
static inline void launch_gemm(const bf16* A, const bf16* W, const float* bias,
                               float* C, int M, int N, int K, hipStream_t s)
{
    int tiles  = (M >> 4) * (N >> 6);       // 16x64 output blocks per wave
    int blocks = (tiles + 7) / 8;           // 8 waves (256 thr) per block
    gemm_bf16_wmma4<<<blocks, 256, 0, s>>>(A, W, bias, C, M, N, K);
}

static inline void launch_cvt(const float* src, bf16* dst, int n, hipStream_t s)
{
    cvt_bf16<<<(n + 255) / 256, 256, 0, s>>>(src, dst, n);
}

extern "C" void kernel_launch(void* const* d_in, const int* in_sizes, int n_in,
                              void* d_out, int out_size, void* d_ws, size_t ws_size,
                              hipStream_t stream)
{
    (void)in_sizes; (void)n_in; (void)out_size; (void)ws_size;

    const int*   x        = (const int*)  d_in[0];
    const float* mp       = (const float*)d_in[1];
    const float* y        = (const float*)d_in[2];
    const float* emb      = (const float*)d_in[3];
    const float* eWih0    = (const float*)d_in[4];
    const float* eWhh0    = (const float*)d_in[5];
    const float* ebih0    = (const float*)d_in[6];
    const float* ebhh0    = (const float*)d_in[7];
    const float* eWih1    = (const float*)d_in[8];
    const float* eWhh1    = (const float*)d_in[9];
    const float* ebih1    = (const float*)d_in[10];
    const float* ebhh1    = (const float*)d_in[11];
    const float* dWih0    = (const float*)d_in[12];
    const float* dWhh0    = (const float*)d_in[13];
    const float* dbih0    = (const float*)d_in[14];
    const float* dbhh0    = (const float*)d_in[15];
    const float* dWih1    = (const float*)d_in[16];
    const float* dWhh1    = (const float*)d_in[17];
    const float* dbih1    = (const float*)d_in[18];
    const float* dbhh1    = (const float*)d_in[19];
    const float* mapW     = (const float*)d_in[20];
    const float* mapB     = (const float*)d_in[21];
    float* out            = (float*)d_out;

    // ---- workspace layout: fp32 region first, then bf16 region ------
    float* w      = (float*)d_ws;
    float* Gi     = w;  w += (size_t)BB * SIN * G3H;     // batched input-gates (reused)
    float* enc    = w;  w += (size_t)BB * SIN * HH;      // encoder output seq (fp32: attention)
    float* gh     = w;  w += (size_t)BB * G3H;           // per-step hidden gates
    float* gi1    = w;  w += (size_t)BB * G3H;           // decoder layer-1 input gates
    float* hA     = w;  w += (size_t)BB * HH;
    float* hB     = w;  w += (size_t)BB * HH;
    float* h0A    = w;  w += (size_t)BB * HH;
    float* h0B    = w;  w += (size_t)BB * HH;
    float* h1A    = w;  w += (size_t)BB * HH;
    float* h1B    = w;  w += (size_t)BB * HH;
    float* scores = w;  w += (size_t)BB * SIN;
    float* alpha  = w;  w += (size_t)BB * SIN;
    float* logits = w;  w += (size_t)BB * YD;

    bf16* bw      = (bf16*)w;
    bf16* xe_bf   = bw; bw += (size_t)BB * SIN * EE;     // embedded input
    bf16* l0_bf   = bw; bw += (size_t)BB * SIN * HH;     // enc layer-0 seq (GEMM A)
    bf16* din_bf  = bw; bw += (size_t)BB * SOUT * DIN;   // decoder inputs
    bf16* ctx_bf  = bw; bw += (size_t)BB * 2 * HH;       // [attn, h1] concat
    bf16* hAb     = bw; bw += (size_t)BB * HH;
    bf16* hBb     = bw; bw += (size_t)BB * HH;
    bf16* h0Ab    = bw; bw += (size_t)BB * HH;
    bf16* h0Bb    = bw; bw += (size_t)BB * HH;
    bf16* h1Ab    = bw; bw += (size_t)BB * HH;
    bf16* h1Bb    = bw; bw += (size_t)BB * HH;
    // bf16 weights (converted once per call)
    bf16* beWih0  = bw; bw += (size_t)G3H * EE;
    bf16* beWhh0  = bw; bw += (size_t)G3H * HH;
    bf16* beWih1  = bw; bw += (size_t)G3H * HH;
    bf16* beWhh1  = bw; bw += (size_t)G3H * HH;
    bf16* bdWih0  = bw; bw += (size_t)G3H * DIN;
    bf16* bdWhh0  = bw; bw += (size_t)G3H * HH;
    bf16* bdWih1  = bw; bw += (size_t)G3H * HH;
    bf16* bdWhh1  = bw; bw += (size_t)G3H * HH;
    bf16* bmapW   = bw; bw += (size_t)YD * 2 * HH;
    float* accum  = (float*)bw;                           // [loss, ysum]

    const int BH = BB * HH;

    // ---- weight conversion (per call; no cross-call state) ----------
    launch_cvt(eWih0, beWih0, G3H * EE,      stream);
    launch_cvt(eWhh0, beWhh0, G3H * HH,      stream);
    launch_cvt(eWih1, beWih1, G3H * HH,      stream);
    launch_cvt(eWhh1, beWhh1, G3H * HH,      stream);
    launch_cvt(dWih0, bdWih0, G3H * DIN,     stream);
    launch_cvt(dWhh0, bdWhh0, G3H * HH,      stream);
    launch_cvt(dWih1, bdWih1, G3H * HH,      stream);
    launch_cvt(dWhh1, bdWhh1, G3H * HH,      stream);
    launch_cvt(mapW,  bmapW,  YD * 2 * HH,   stream);

    // ---- init accumulators / initial hidden states ------------------
    zero_kernel<<<(BH + 255) / 256, 256, 0, stream>>>(hA, BH);
    zero_kernel<<<(BH + 255) / 256, 256, 0, stream>>>(h0A, BH);
    zero_kernel<<<(BH + 255) / 256, 256, 0, stream>>>(h1A, BH);
    zero_bf16<<<(BH + 255) / 256, 256, 0, stream>>>(hAb, BH);
    zero_bf16<<<(BH + 255) / 256, 256, 0, stream>>>(h0Ab, BH);
    zero_bf16<<<(BH + 255) / 256, 256, 0, stream>>>(h1Ab, BH);
    zero_kernel<<<1, 256, 0, stream>>>(accum, 2);
    sum_all<<<256, 256, 0, stream>>>(y, BB * SOUT * YD, accum + 1);

    // ---- embedding (bf16 out) ---------------------------------------
    {
        int n = BB * SIN * EE;
        embed_kernel<<<(n + 255) / 256, 256, 0, stream>>>(x, emb, xe_bf);
    }

    // ---- encoder layer 0 --------------------------------------------
    launch_gemm(xe_bf, beWih0, ebih0, Gi, BB * SIN, G3H, EE, stream);
    {
        float* hc = hA;  float* hn = hB;
        bf16*  hcb = hAb; bf16* hnb = hBb;
        for (int t = 0; t < SIN; ++t) {
            launch_gemm(hcb, beWhh0, ebhh0, gh, BB, G3H, HH, stream);
            gru_update<<<(BH + 255) / 256, 256, 0, stream>>>(
                Gi + (size_t)t * G3H, SIN * G3H, gh, hc, hn, hnb,
                (float*)nullptr, l0_bf + (size_t)t * HH, SIN * HH);
            float* tf = hc; hc = hn; hn = tf;
            bf16*  tb = hcb; hcb = hnb; hnb = tb;
        }
    }

    // ---- encoder layer 1 --------------------------------------------
    zero_kernel<<<(BH + 255) / 256, 256, 0, stream>>>(hA, BH);
    zero_bf16<<<(BH + 255) / 256, 256, 0, stream>>>(hAb, BH);
    launch_gemm(l0_bf, beWih1, ebih1, Gi, BB * SIN, G3H, HH, stream);
    {
        float* hc = hA;  float* hn = hB;
        bf16*  hcb = hAb; bf16* hnb = hBb;
        for (int t = 0; t < SIN; ++t) {
            launch_gemm(hcb, beWhh1, ebhh1, gh, BB, G3H, HH, stream);
            gru_update<<<(BH + 255) / 256, 256, 0, stream>>>(
                Gi + (size_t)t * G3H, SIN * G3H, gh, hc, hn, hnb,
                enc + (size_t)t * HH, (bf16*)nullptr, SIN * HH);
            float* tf = hc; hc = hn; hn = tf;
            bf16*  tb = hcb; hcb = hnb; hnb = tb;
        }
    }

    // ---- decoder ----------------------------------------------------
    {
        int n = BB * SOUT * DIN;
        build_dec_in<<<(n + 255) / 256, 256, 0, stream>>>(mp, y, din_bf);
    }
    launch_gemm(din_bf, bdWih0, dbih0, Gi, BB * SOUT, G3H, DIN, stream);

    {
        float* h0c = h0A;  float* h0n = h0B;
        bf16*  h0cb = h0Ab; bf16* h0nb = h0Bb;
        float* h1c = h1A;  float* h1n = h1B;
        bf16*  h1cb = h1Ab; bf16* h1nb = h1Bb;
        for (int t = 0; t < SOUT; ++t) {
            // layer 0
            launch_gemm(h0cb, bdWhh0, dbhh0, gh, BB, G3H, HH, stream);
            gru_update<<<(BH + 255) / 256, 256, 0, stream>>>(
                Gi + (size_t)t * G3H, SOUT * G3H, gh, h0c, h0n, h0nb,
                (float*)nullptr, (bf16*)nullptr, 0);
            // layer 1
            launch_gemm(h0nb, bdWih1, dbih1, gi1, BB, G3H, HH, stream);
            launch_gemm(h1cb, bdWhh1, dbhh1, gh,  BB, G3H, HH, stream);
            gru_update<<<(BH + 255) / 256, 256, 0, stream>>>(
                gi1, G3H, gh, h1c, h1n, h1nb,
                (float*)nullptr, (bf16*)nullptr, 0);
            // attention over encoder outputs
            attn_scores<<<(BB * SIN + 7) / 8, 256, 0, stream>>>(enc, h1n, scores);
            softmax128<<<BB, SIN, 0, stream>>>(scores, alpha);
            attn_apply<<<(BH + 255) / 256, 256, 0, stream>>>(enc, alpha, h1n, ctx_bf);
            // projection + loss
            launch_gemm(ctx_bf, bmapW, mapB, logits, BB, YD, 2 * HH, stream);
            loss_kernel<<<BB, YD, 0, stream>>>(logits, y, t, accum);
            // rotate states
            float* tf; bf16* tb;
            tf = h0c; h0c = h0n; h0n = tf;   tb = h0cb; h0cb = h0nb; h0nb = tb;
            tf = h1c; h1c = h1n; h1n = tf;   tb = h1cb; h1cb = h1nb; h1nb = tb;
        }
    }

    finalize<<<1, 32, 0, stream>>>(accum, accum + 1, out);
}
